// LipikaRVQAdapter_21483426415071
// MI455X (gfx1250) — matrix operand order repaired and checked by reading, not compiled, not run on hardware
//
#include <hip/hip_runtime.h>

typedef __attribute__((ext_vector_type(16))) __bf16 v16bf;
typedef __attribute__((ext_vector_type(8)))  float  v8f;
typedef int v4i __attribute__((vector_size(16)));

#define NCB   8
#define KCB   1024
#define CDIM  128
#define ENC   512
#define INDIM 768
#define TOK   32768   // 16 * 2048
#define MROWS 64      // tokens per block (4 WMMA M-tiles)

// LDS row strides padded so lane-m b128 loads hit distinct banks
#define ZT_LD    776   // 768 + 8
#define RES_LD   520   // 512 + 8
#define ZE_LD    136   // 128 + 8

// keep per-iteration LDS fragment loads inside their loop (stop LICM from
// hoisting loop-invariant A-fragments and spilling them to scratch)
#define NO_HOIST() asm volatile("" ::: "memory")

#if defined(__has_builtin)
#if __has_builtin(__builtin_amdgcn_global_load_async_to_lds_b128) && \
    __has_builtin(__builtin_amdgcn_s_wait_asynccnt)
#define USE_ASYNC_LDS 1
typedef __attribute__((address_space(1))) v4i* gv4i_p;   // global
typedef __attribute__((address_space(3))) v4i* lv4i_p;   // LDS
#endif
#endif

__device__ __forceinline__ v16bf ld_frag(const __bf16* p0, const __bf16* p1) {
  v16bf r;
  ((uint4*)&r)[0] = *(const uint4*)p0;
  ((uint4*)&r)[1] = *(const uint4*)p1;
  return r;
}

__device__ __forceinline__ v8f wmma_bf16(v16bf a, v16bf b, v8f c) {
  return __builtin_amdgcn_wmma_f32_16x16x32_bf16(false, a, false, b, (short)0, c, false, false);
}

// ---------------- weight prep: fp32 -> bf16, transposed for B-fragment loads ----------------
__global__ void prep_weights(const float* __restrict__ W_in, const float* __restrict__ cbs,
                             const float* __restrict__ in_w, const float* __restrict__ out_w,
                             __bf16* __restrict__ winT, __bf16* __restrict__ inwT,
                             __bf16* __restrict__ outwT, __bf16* __restrict__ cbbf) {
  long idx = (long)blockIdx.x * blockDim.x + threadIdx.x;
  const long nWin = (long)ENC * INDIM;          // winT[c][f]
  const long nIn  = (long)NCB * CDIM * ENC;     // inwT[k][c][d]
  const long nOut = (long)NCB * ENC * CDIM;     // outwT[k][d][c]
  const long nCb  = (long)NCB * KCB * CDIM;     // cbbf as-is
  if (idx < nWin) {
    long c = idx / INDIM, f = idx % INDIM;
    winT[idx] = (__bf16)W_in[f * ENC + c];
  } else if ((idx -= nWin) < nIn) {
    long k = idx / (CDIM * ENC), r = idx % (CDIM * ENC);
    long c = r / ENC, d = r % ENC;
    inwT[k * CDIM * ENC + c * ENC + d] = (__bf16)in_w[k * ENC * CDIM + d * CDIM + c];
  } else if ((idx -= nIn) < nOut) {
    long k = idx / (ENC * CDIM), r = idx % (ENC * CDIM);
    long d = r / CDIM, c = r % CDIM;
    outwT[k * ENC * CDIM + d * CDIM + c] = (__bf16)out_w[k * CDIM * ENC + c * ENC + d];
  } else if ((idx -= nOut) < nCb) {
    cbbf[idx] = (__bf16)cbs[idx];
  }
}

__global__ void prep_norms(const float* __restrict__ cbs, float* __restrict__ cbn) {
  int e = blockIdx.x * blockDim.x + threadIdx.x;     // 8192 entries
  if (e >= NCB * KCB) return;
  const float* row = cbs + (long)e * CDIM;
  float s = 0.f;
  #pragma unroll 8
  for (int c = 0; c < CDIM; ++c) { float v = row[c]; s += v * v; }
  cbn[e] = 0.5f * s;
}

// ---------------- projection: z(64x768) @ W_in -> z_enc bf16 ----------------
__global__ void __launch_bounds__(256)
proj_kernel(const float* __restrict__ z, const __bf16* __restrict__ winT,
            const float* __restrict__ bias, __bf16* __restrict__ zenc) {
  __shared__ alignas(16) __bf16 zt[MROWS * ZT_LD];
  const int tid = threadIdx.x;
  const int wave = tid >> 5, lane = tid & 31;
  const int half = lane >> 4, l15 = lane & 15;
  const long r0 = (long)blockIdx.x * MROWS;

  for (int i = tid; i < MROWS * INDIM; i += 256) {
    int row = i / INDIM, c = i % INDIM;
    zt[row * ZT_LD + c] = (__bf16)z[(r0 + row) * INDIM + c];
  }
  __syncthreads();

  for (int t = 0; t < 4; ++t) {
    NO_HOIST();
    const int col = t * 128 + wave * 16;
    v8f acc[4] = {};
    const __bf16* bbase = winT + (long)(col + l15) * INDIM;
    #pragma unroll
    for (int kk = 0; kk < INDIM / 32; ++kk) {
      const __bf16* bp = bbase + kk * 32 + 8 * half;
      v16bf b = ld_frag(bp, bp + 16);           // one B fragment ...
      #pragma unroll
      for (int mt = 0; mt < 4; ++mt) {          // ... reused over 4 M-tiles
        const __bf16* ap = zt + (mt * 16 + l15) * ZT_LD + kk * 32 + 8 * half;
        acc[mt] = wmma_bf16(ld_frag(ap, ap + 16), b, acc[mt]);
      }
    }
    const float bv = bias[col + l15];
    #pragma unroll
    for (int mt = 0; mt < 4; ++mt)
      #pragma unroll
      for (int i = 0; i < 8; ++i) {
        int m = mt * 16 + i + 8 * half;
        zenc[(r0 + m) * ENC + col + l15] = (__bf16)(acc[mt][i] + bv);
      }
  }
}

// ---------------- residual VQ: 8 codebooks, 64 tokens per block ----------------
__global__ void __launch_bounds__(256)
rvq_kernel(const __bf16* __restrict__ zenc, const __bf16* __restrict__ inwT,
           const __bf16* __restrict__ outwT, const __bf16* __restrict__ cbbf,
           const float* __restrict__ cbn, float* __restrict__ codes,
           float* __restrict__ loss) {
  __shared__ alignas(16) __bf16 resid[MROWS * RES_LD];
  __shared__ alignas(16) __bf16 ze[MROWS * ZE_LD];
  __shared__ alignas(16) __bf16 zq[MROWS * ZE_LD];
  __shared__ float cand_s[8 * MROWS];
  __shared__ int   cand_i[8 * MROWS];
  __shared__ int   sel[MROWS];

  const int tid = threadIdx.x;
  const int wave = tid >> 5, lane = tid & 31;
  const int half = lane >> 4, l15 = lane & 15;
  const long r0 = (long)blockIdx.x * MROWS;

  // stage residual = z_enc tile (bf16, padded rows) via ASYNCcnt path when available
  {
    const uint4* src = (const uint4*)(zenc + r0 * ENC);   // 4096 uint4
    for (int i = tid; i < MROWS * (ENC / 8); i += 256) {
      int row = i >> 6, p = i & 63;
      uint4* dst = (uint4*)(resid + row * RES_LD) + p;
#ifdef USE_ASYNC_LDS
      __builtin_amdgcn_global_load_async_to_lds_b128(
          (gv4i_p)(src + i), (lv4i_p)dst, 0, 0);
#else
      *dst = src[i];
#endif
    }
#ifdef USE_ASYNC_LDS
    __builtin_amdgcn_s_wait_asynccnt(0);
#endif
  }
  __syncthreads();

  float lacc = 0.f;

  for (int k = 0; k < NCB; ++k) {
    // pull next codebook's operand streams toward the WGP while this step computes
    if (k + 1 < NCB) {
      __builtin_prefetch((const char*)(cbbf + (long)(k + 1) * KCB * CDIM) + tid * 1024, 0, 1);
      __builtin_prefetch((const char*)(inwT + (long)(k + 1) * CDIM * ENC) + tid * 512, 0, 1);
      __builtin_prefetch((const char*)(outwT + (long)(k + 1) * ENC * CDIM) + tid * 512, 0, 1);
    }

    // ---- 1: z_e = resid @ in_w[k]   (64x512 @ 512x128), one 16-col tile per wave
    {
      const int col = wave * 16;
      v8f acc[4] = {};
      const __bf16* bbase = inwT + (long)k * CDIM * ENC + (long)(col + l15) * ENC;
      #pragma unroll
      for (int kk = 0; kk < ENC / 32; ++kk) {
        const __bf16* bp = bbase + kk * 32 + 8 * half;
        v16bf b = ld_frag(bp, bp + 16);
        #pragma unroll
        for (int mt = 0; mt < 4; ++mt) {
          const __bf16* ap = resid + (mt * 16 + l15) * RES_LD + kk * 32 + 8 * half;
          acc[mt] = wmma_bf16(ld_frag(ap, ap + 16), b, acc[mt]);
        }
      }
      #pragma unroll
      for (int mt = 0; mt < 4; ++mt)
        #pragma unroll
        for (int i = 0; i < 8; ++i)
          ze[(mt * 16 + i + 8 * half) * ZE_LD + col + l15] = (__bf16)acc[mt][i];
    }
    __syncthreads();

    // ---- 2: scores = z_e @ cb^T, running argmax of (dot - 0.5||cb||^2); 128 entries/wave
    float bs[4][8]; int bi[4][8];
    #pragma unroll
    for (int mt = 0; mt < 4; ++mt)
      #pragma unroll
      for (int i = 0; i < 8; ++i) { bs[mt][i] = -3.4e38f; bi[mt][i] = 0; }

    for (int t = 0; t < 8; ++t) {
      NO_HOIST();
      const int e0 = wave * 128 + t * 16;
      v8f s[4] = {};
      const __bf16* bbase = cbbf + ((long)k * KCB + e0 + l15) * CDIM;
      #pragma unroll
      for (int kk = 0; kk < CDIM / 32; ++kk) {
        const __bf16* bp = bbase + kk * 32 + 8 * half;
        v16bf b = ld_frag(bp, bp + 16);
        #pragma unroll
        for (int mt = 0; mt < 4; ++mt) {
          const __bf16* ap = ze + (mt * 16 + l15) * ZE_LD + kk * 32 + 8 * half;
          s[mt] = wmma_bf16(ld_frag(ap, ap + 16), b, s[mt]);
        }
      }
      const float cn = cbn[k * KCB + e0 + l15];
      const int   ce = e0 + l15;
      #pragma unroll
      for (int mt = 0; mt < 4; ++mt)
        #pragma unroll
        for (int i = 0; i < 8; ++i) {
          float sc = s[mt][i] - cn;
          if (sc > bs[mt][i]) { bs[mt][i] = sc; bi[mt][i] = ce; }
        }
    }
    // reduce across the 16 lanes holding different entries (each wave32 half)
    #pragma unroll
    for (int mt = 0; mt < 4; ++mt)
      #pragma unroll
      for (int i = 0; i < 8; ++i) {
        float v = bs[mt][i]; int ix = bi[mt][i];
        #pragma unroll
        for (int m = 8; m >= 1; m >>= 1) {
          float ov = __shfl_xor(v, m, 32);
          int   oi = __shfl_xor(ix, m, 32);
          if (ov > v || (ov == v && oi < ix)) { v = ov; ix = oi; }
        }
        bs[mt][i] = v; bi[mt][i] = ix;
      }
    if (l15 == 0) {                       // lanes 0 and 16
      #pragma unroll
      for (int mt = 0; mt < 4; ++mt)
        #pragma unroll
        for (int i = 0; i < 8; ++i) {
          int m = mt * 16 + i + 8 * half;
          cand_s[wave * MROWS + m] = bs[mt][i];
          cand_i[wave * MROWS + m] = bi[mt][i];
        }
    }
    __syncthreads();
    if (tid < MROWS) {
      float v = cand_s[tid]; int ix = cand_i[tid];
      #pragma unroll
      for (int w = 1; w < 8; ++w) {
        float ov = cand_s[w * MROWS + tid]; int oi = cand_i[w * MROWS + tid];
        if (ov > v || (ov == v && oi < ix)) { v = ov; ix = oi; }
      }
      sel[tid] = ix;
      codes[(r0 + tid) * NCB + k] = (float)ix;
    }
    __syncthreads();

    // ---- 3: gather z_q = cb[idx] + commitment/codebook loss
    #pragma unroll
    for (int j = 0; j < 4; ++j) {
      int idx = j * 256 + tid;            // 1024 uint4 = 64 rows x 16 chunks
      int row = idx >> 4, part = idx & 15;
      const uint4* src = (const uint4*)(cbbf + ((long)k * KCB + sel[row]) * CDIM) + part;
      uint4 q = *src;
      ((uint4*)(zq + row * ZE_LD))[part] = q;
      const __bf16* qb  = (const __bf16*)&q;
      const __bf16* zeb = ze + row * ZE_LD + part * 8;
      #pragma unroll
      for (int i = 0; i < 8; ++i) {
        float d = (float)zeb[i] - (float)qb[i];
        lacc += d * d;
      }
    }
    __syncthreads();

    // ---- 4: dec = z_q @ out_w[k]; resid -= dec
    for (int t = 0; t < 4; ++t) {
      NO_HOIST();
      const int col = t * 128 + wave * 16;
      v8f dacc[4] = {};
      const __bf16* bbase = outwT + (long)k * ENC * CDIM + (long)(col + l15) * CDIM;
      #pragma unroll
      for (int kk = 0; kk < CDIM / 32; ++kk) {
        const __bf16* bp = bbase + kk * 32 + 8 * half;
        v16bf b = ld_frag(bp, bp + 16);
        #pragma unroll
        for (int mt = 0; mt < 4; ++mt) {
          const __bf16* ap = zq + (mt * 16 + l15) * ZE_LD + kk * 32 + 8 * half;
          dacc[mt] = wmma_bf16(ld_frag(ap, ap + 16), b, dacc[mt]);
        }
      }
      #pragma unroll
      for (int mt = 0; mt < 4; ++mt)
        #pragma unroll
        for (int i = 0; i < 8; ++i) {
          int off = (mt * 16 + i + 8 * half) * RES_LD + col + l15;
          resid[off] = (__bf16)((float)resid[off] - dacc[mt][i]);
        }
    }
    __syncthreads();
  }

  // loss: (cb + 0.25*commit) both equal mean((ze - zq)^2) numerically
  lacc *= 1.25f / ((float)TOK * (float)CDIM);
  #pragma unroll
  for (int m = 16; m >= 1; m >>= 1) lacc += __shfl_xor(lacc, m, 32);
  if (lane == 0) atomicAdd(loss, lacc);
}

extern "C" void kernel_launch(void* const* d_in, const int* in_sizes, int n_in,
                              void* d_out, int out_size, void* d_ws, size_t ws_size,
                              hipStream_t stream) {
  const float* z    = (const float*)d_in[0];
  const float* W_in = (const float*)d_in[1];
  const float* b_in = (const float*)d_in[2];
  const float* cbs  = (const float*)d_in[3];
  const float* in_w = (const float*)d_in[4];
  const float* outw = (const float*)d_in[5];

  char* ws = (char*)d_ws;
  __bf16* winT  = (__bf16*)(ws + 0);                    // 512*768*2   = 786432
  __bf16* inwT  = (__bf16*)(ws + 786432);               // 8*128*512*2 = 1048576
  __bf16* outwT = (__bf16*)(ws + 1835008);              // 8*512*128*2 = 1048576
  __bf16* cbbf  = (__bf16*)(ws + 2883584);              // 8*1024*128*2= 2097152
  float*  cbn   = (float*)(ws + 4980736);               // 8192*4      = 32768
  __bf16* zenc  = (__bf16*)(ws + 5013504);              // 32768*512*2 = 33554432

  float* codes = (float*)d_out;                         // 16*2048*8 entries
  float* loss  = (float*)d_out + (long)TOK * NCB;       // final scalar

  (void)hipMemsetAsync(loss, 0, sizeof(float), stream);

  {
    long total = (long)ENC * INDIM + 2L * NCB * CDIM * ENC + (long)NCB * KCB * CDIM;
    int blocks = (int)((total + 255) / 256);
    prep_weights<<<blocks, 256, 0, stream>>>(W_in, cbs, in_w, outw, winT, inwT, outwT, cbbf);
  }
  prep_norms<<<(NCB * KCB + 255) / 256, 256, 0, stream>>>(cbs, cbn);
  proj_kernel<<<TOK / MROWS, 256, 0, stream>>>(z, winT, b_in, zenc);
  rvq_kernel<<<TOK / MROWS, 256, 0, stream>>>(zenc, inwT, outwT, cbbf, cbn, codes, loss);
}
